// MultiHeadAttention_798863917768
// MI455X (gfx1250) — matrix operand order repaired and checked by reading, not compiled, not run on hardware
//
#include <hip/hip_runtime.h>
#include <hip/hip_bf16.h>
#include <cstdint>

// MHA forward for MI455X (gfx1250): B=2, L=2048, E=1024, H=16, D=64.
// All matmuls via v_wmma_f32_16x16x32_f16 (f16 in, f32 accumulate).

#define B_   2
#define L_   2048
#define E_   1024
#define H_   16
#define D_   64
#define F_   3072       // 3*H*D
#define BL_  4096       // B*L
#define HD_  1024       // H*D

typedef __attribute__((ext_vector_type(16))) _Float16 v16h;
typedef __attribute__((ext_vector_type(8)))  float    v8f;
typedef __attribute__((ext_vector_type(4)))  uint32_t v4u;
typedef __attribute__((ext_vector_type(8)))  uint32_t v8u;

static_assert(sizeof(v16h) == 32, "v16h must be 32 bytes");

__device__ __forceinline__ v16h make_frag(v4u lo, v4u hi) {
    v8u t;
    t[0] = lo[0]; t[1] = lo[1]; t[2] = lo[2]; t[3] = lo[3];
    t[4] = hi[0]; t[5] = hi[1]; t[6] = hi[2]; t[7] = hi[3];
    return __builtin_bit_cast(v16h, t);
}

// A fragment (16x32 f16, M x K): base points at row m0, col k0 of a row-major
// matrix with leading dim ld. ISA layout: lane m = lane&15, g = lane>>4;
// elems 0..7 -> k = 8g..8g+7, elems 8..15 -> k = 16+8g..16+8g+7.
__device__ __forceinline__ v16h load_a(const _Float16* __restrict__ base, int ld, int lane) {
    const int m = lane & 15, g = lane >> 4;
    const _Float16* p = base + (size_t)m * ld + 8 * g;
    v4u lo = *(const v4u*)(p);
    v4u hi = *(const v4u*)(p + 16);
    return make_frag(lo, hi);
}

// B fragment (32x16 f16, K x N) where B[k][n] = W[n][k], W row-major with
// leading dim ld (reduction dim contiguous). base points at row n0, col k0.
// ISA layout: lane n = lane&15, g = lane>>4; elems 0..7 -> k = 8g..8g+7, elems 8..15 -> k = 16+8g..16+8g+7.
__device__ __forceinline__ v16h load_bT(const _Float16* __restrict__ base, int ld, int lane) {
    const int n = lane & 15, g = lane >> 4;
    const _Float16* p = base + (size_t)n * ld + 8 * g;
    v4u lo = *(const v4u*)(p);
    v4u hi = *(const v4u*)(p + 16);
    return make_frag(lo, hi);
}

__device__ __forceinline__ v8f wmma16(v16h a, v16h b, v8f c) {
    return __builtin_amdgcn_wmma_f32_16x16x32_f16(false, a, false, b, (short)0, c, false, false);
}

// ---------------------------------------------------------------- convert
__global__ void cvt_f32_f16(const float* __restrict__ in, _Float16* __restrict__ out, int n) {
    int i = blockIdx.x * blockDim.x + threadIdx.x;
    if (i < n) out[i] = (_Float16)in[i];
}

// ---------------------------------------------------------------- QKV GEMM
// tmp[m][f] = sum_e x[m][e] * Wa[f][e];  m in [0,4096), f in [0,3072)
// Scatter-store into Q [B,H,L,D], K [B,H,L,D], Vt [B,H,D,L] (V transposed).
__global__ __launch_bounds__(256) void qkv_gemm(const _Float16* __restrict__ xh,
                                                const _Float16* __restrict__ wah,
                                                _Float16* __restrict__ Qb,
                                                _Float16* __restrict__ Kb,
                                                _Float16* __restrict__ Vt) {
    const int lane = threadIdx.x & 31;
    const int wave = (blockIdx.x * blockDim.x + threadIdx.x) >> 5;
    const int MT = BL_ / 32;   // 128
    const int NT = F_ / 64;    // 48
    if (wave >= MT * NT) return;
    const int mt = wave % MT, nt = wave / MT;
    const int row0 = mt * 32, col0 = nt * 64;

    v8f zero = {};
    v8f acc[2][4];
    for (int i = 0; i < 2; ++i)
        for (int j = 0; j < 4; ++j) acc[i][j] = zero;

    for (int k0 = 0; k0 < E_; k0 += 32) {
        v16h a0 = load_a(xh + (size_t)(row0     ) * E_ + k0, E_, lane);
        v16h a1 = load_a(xh + (size_t)(row0 + 16) * E_ + k0, E_, lane);
        for (int j = 0; j < 4; ++j) {
            v16h bf = load_bT(wah + (size_t)(col0 + 16 * j) * E_ + k0, E_, lane);
            acc[0][j] = wmma16(a0, bf, acc[0][j]);
            acc[1][j] = wmma16(a1, bf, acc[1][j]);
        }
    }

    const int nloc = lane & 15, g = lane >> 4;
    for (int j = 0; j < 4; ++j) {
        const int f0 = col0 + 16 * j;
        const int h  = f0 / 192;
        const int r0 = f0 % 192;   // multiple of 16; segment-pure tile
        for (int i = 0; i < 2; ++i) {
            for (int r = 0; r < 8; ++r) {
                const int row = row0 + 16 * i + r + 8 * g;   // global (b,l)
                const int b = row >> 11;                      // /2048
                const int l = row & 2047;
                const _Float16 val = (_Float16)acc[i][j][r];
                if (r0 < 64) {
                    const int d = r0 + nloc;
                    Qb[(((size_t)(b * H_ + h)) * L_ + l) * D_ + d] = val;
                } else if (r0 < 128) {
                    const int d = r0 - 64 + nloc;
                    Kb[(((size_t)(b * H_ + h)) * L_ + l) * D_ + d] = val;
                } else {
                    const int d = r0 - 128 + nloc;
                    Vt[(((size_t)(b * H_ + h)) * D_ + d) * L_ + l] = val;
                }
            }
        }
    }
}

// ---------------------------------------------------------------- attention
// One wave per (b, h, 16-query tile). Flash-style online softmax, 32 keys/step.
__global__ __launch_bounds__(256) void attn_fwd(const _Float16* __restrict__ Qb,
                                                const _Float16* __restrict__ Kb,
                                                const _Float16* __restrict__ Vt,
                                                _Float16* __restrict__ Ob) {
    __shared__ __align__(16) float    Sbuf[8][16][32];
    __shared__ __align__(16) _Float16 Pbuf[8][16][32];
    __shared__               float    Abuf[8][16];

    const int lane = threadIdx.x & 31;
    const int wv   = threadIdx.x >> 5;
    const int wave = (blockIdx.x * blockDim.x + threadIdx.x) >> 5;
    const int QT = L_ / 16;                 // 128
    if (wave >= B_ * H_ * QT) return;
    const int qt = wave % QT;
    const int bh = wave / QT;
    const int q0 = qt * 16;

    const _Float16* Qp = Qb + (size_t)bh * L_ * D_;
    const _Float16* Kp = Kb + (size_t)bh * L_ * D_;
    const _Float16* Vp = Vt + (size_t)bh * D_ * L_;

    const int nloc = lane & 15, g = lane >> 4;

    // Q fragments for this 16-row tile (d = 0..31, 32..63)
    const v16h qf0 = load_a(Qp + (size_t)q0 * D_ +  0, D_, lane);
    const v16h qf1 = load_a(Qp + (size_t)q0 * D_ + 32, D_, lane);

    v8f zero = {};
    v8f acc[4];
    for (int t = 0; t < 4; ++t) acc[t] = zero;
    float m_i = -1e30f, l_i = 0.0f;
    const float scale = 0.125f;   // 1/sqrt(64)

    for (int j0 = 0; j0 <= q0 + 15; j0 += 32) {
        // S = Q K^T for keys [j0, j0+31], two 16x16 tiles -> LDS (masked+scaled)
        for (int bt = 0; bt < 2; ++bt) {
            v16h kb0 = load_bT(Kp + (size_t)(j0 + 16 * bt) * D_ +  0, D_, lane);
            v16h kb1 = load_bT(Kp + (size_t)(j0 + 16 * bt) * D_ + 32, D_, lane);
            v8f s = zero;
            s = wmma16(qf0, kb0, s);
            s = wmma16(qf1, kb1, s);
            const int key = j0 + 16 * bt + nloc;
            for (int r = 0; r < 8; ++r) {
                const int qrow = q0 + r + 8 * g;
                float sv = s[r] * scale;
                if (key > qrow) sv = -1e30f;
                Sbuf[wv][r + 8 * g][16 * bt + nloc] = sv;
            }
        }
        // lane owns row = nloc, keys [16g, 16g+15]; pair (lane, lane^16) = full row
        float sv[16];
        for (int e = 0; e < 16; ++e) sv[e] = Sbuf[wv][nloc][16 * g + e];
        float tmax = sv[0];
        for (int e = 1; e < 16; ++e) tmax = fmaxf(tmax, sv[e]);
        tmax = fmaxf(tmax, __shfl_xor(tmax, 16, 32));
        const float m_new = fmaxf(m_i, tmax);
        const float alpha = __expf(m_i - m_new);
        float psum = 0.0f;
        for (int e = 0; e < 16; ++e) {
            const float p = __expf(sv[e] - m_new);
            psum += p;
            Pbuf[wv][nloc][16 * g + e] = (_Float16)p;
        }
        psum += __shfl_xor(psum, 16, 32);
        l_i = alpha * l_i + psum;
        m_i = m_new;
        if (g == 0) Abuf[wv][nloc] = alpha;

        // rescale running O accumulators (alpha for row r+8g)
        float ar[8];
        for (int r = 0; r < 8; ++r) ar[r] = Abuf[wv][r + 8 * g];
        for (int t = 0; t < 4; ++t)
            for (int r = 0; r < 8; ++r) acc[t][r] *= ar[r];

        // P fragment (A layout, 16x32) straight out of LDS
        const _Float16* pp = &Pbuf[wv][nloc][0];
        v4u plo = *(const v4u*)(pp + 8 * g);
        v4u phi = *(const v4u*)(pp + 16 + 8 * g);
        const v16h pf = make_frag(plo, phi);

        // O += P * V ; V pre-transposed [D][L] so B-frags are contiguous loads
        for (int t = 0; t < 4; ++t) {
            v16h vf = load_bT(Vp + (size_t)(16 * t) * L_ + j0, L_, lane);
            acc[t] = wmma16(pf, vf, acc[t]);
        }
    }

    // finalize: divide by running sum, write o as [B*L][H*D] f16
    if (g == 0) Abuf[wv][nloc] = l_i;
    float lr[8];
    for (int r = 0; r < 8; ++r) lr[r] = 1.0f / Abuf[wv][r + 8 * g];

    const int b = bh / H_, h = bh % H_;
    for (int t = 0; t < 4; ++t) {
        for (int r = 0; r < 8; ++r) {
            const int l = q0 + r + 8 * g;
            const int d = 16 * t + nloc;
            Ob[((size_t)(b * L_ + l)) * HD_ + h * D_ + d] = (_Float16)(acc[t][r] * lr[r]);
        }
    }
}

// ---------------------------------------------------------------- out proj
// out[m][o] = sum_f o[m][f] * Wout[o][f] + bias[o]
__global__ __launch_bounds__(256) void out_gemm(const _Float16* __restrict__ oh,
                                                const _Float16* __restrict__ wh,
                                                const float* __restrict__ bias,
                                                float* __restrict__ out) {
    const int lane = threadIdx.x & 31;
    const int wave = (blockIdx.x * blockDim.x + threadIdx.x) >> 5;
    const int MT = BL_ / 32;   // 128
    const int NT = E_ / 64;    // 16
    if (wave >= MT * NT) return;
    const int mt = wave % MT, nt = wave / MT;
    const int row0 = mt * 32, col0 = nt * 64;

    v8f zero = {};
    v8f acc[2][4];
    for (int i = 0; i < 2; ++i)
        for (int j = 0; j < 4; ++j) acc[i][j] = zero;

    for (int k0 = 0; k0 < HD_; k0 += 32) {
        v16h a0 = load_a(oh + (size_t)(row0     ) * HD_ + k0, HD_, lane);
        v16h a1 = load_a(oh + (size_t)(row0 + 16) * HD_ + k0, HD_, lane);
        for (int j = 0; j < 4; ++j) {
            v16h bf = load_bT(wh + (size_t)(col0 + 16 * j) * HD_ + k0, HD_, lane);
            acc[0][j] = wmma16(a0, bf, acc[0][j]);
            acc[1][j] = wmma16(a1, bf, acc[1][j]);
        }
    }

    const int nloc = lane & 15, g = lane >> 4;
    for (int j = 0; j < 4; ++j) {
        const float bv = bias[col0 + 16 * j + nloc];
        for (int i = 0; i < 2; ++i) {
            for (int r = 0; r < 8; ++r) {
                const int row = row0 + 16 * i + r + 8 * g;
                out[(size_t)row * E_ + col0 + 16 * j + nloc] = acc[i][j][r] + bv;
            }
        }
    }
}

// ---------------------------------------------------------------- launcher
extern "C" void kernel_launch(void* const* d_in, const int* in_sizes, int n_in,
                              void* d_out, int out_size, void* d_ws, size_t ws_size,
                              hipStream_t stream) {
    const float* x   = (const float*)d_in[0];   // [B,L,E]
    const float* Wa  = (const float*)d_in[1];   // [3HD,E]
    const float* Ww  = (const float*)d_in[2];   // [E,HD]
    const float* Wb  = (const float*)d_in[3];   // [E]
    float* out = (float*)d_out;

    uint8_t* ws = (uint8_t*)d_ws;
    // workspace layout (f16 buffers), 48 MB total
    _Float16* xh  = (_Float16*)(ws + 0);              // 4096*1024   = 8 MB
    _Float16* wah = (_Float16*)(ws + 8388608);        // 3072*1024   = 6 MB
    _Float16* wwh = (_Float16*)(ws + 14680064);       // 1024*1024   = 2 MB
    _Float16* Qb  = (_Float16*)(ws + 16777216);       // B*H*L*D     = 8 MB
    _Float16* Kb  = (_Float16*)(ws + 25165824);       // B*H*L*D     = 8 MB
    _Float16* Vt  = (_Float16*)(ws + 33554432);       // B*H*D*L     = 8 MB
    _Float16* Ob  = (_Float16*)(ws + 41943040);       // BL*HD       = 8 MB

    const int n_x = BL_ * E_, n_wa = F_ * E_, n_ww = E_ * HD_;
    cvt_f32_f16<<<(n_x  + 255) / 256, 256, 0, stream>>>(x,  xh,  n_x);
    cvt_f32_f16<<<(n_wa + 255) / 256, 256, 0, stream>>>(Wa, wah, n_wa);
    cvt_f32_f16<<<(n_ww + 255) / 256, 256, 0, stream>>>(Ww, wwh, n_ww);

    // QKV: 128*48 = 6144 waves, 8 waves/block
    qkv_gemm<<<768, 256, 0, stream>>>(xh, wah, Qb, Kb, Vt);
    // attention: 2*16*128 = 4096 waves
    attn_fwd<<<512, 256, 0, stream>>>(Qb, Kb, Vt, Ob);
    // output projection: 128*16 = 2048 waves
    out_gemm<<<256, 256, 0, stream>>>(Ob, wwh, Wb, out);
}
